// MorphingGNN_11811160064841
// MI455X (gfx1250) — compile-verified, matrix operand order checked by reading, not run on hardware
//
#include <hip/hip_runtime.h>
#include <hip/hip_bf16.h>
#include <math.h>

typedef __attribute__((ext_vector_type(16))) _Float16 v16h;
typedef __attribute__((ext_vector_type(8)))  float    v8f;

#define WMMA_F16(a, b, c) \
    __builtin_amdgcn_wmma_f32_16x16x32_f16(false, (a), false, (b), (short)0, (c), false, false)

// ---------------------------------------------------------------------------
// Fragment loaders (CDNA5 wave32 WMMA layouts, ISA 7.12.2)
// A (16x32 f16): lane L -> M = L&15, group = L>>4; half h -> K = (h/8)*16 + group*8 + (h%8)
// caller passes pointer to A_row + kb + group*8
// ---------------------------------------------------------------------------
__device__ __forceinline__ v16h load_a_frag(const float* __restrict__ p) {
    float4 f0 = *reinterpret_cast<const float4*>(p);
    float4 f1 = *reinterpret_cast<const float4*>(p + 4);
    float4 f2 = *reinterpret_cast<const float4*>(p + 16);
    float4 f3 = *reinterpret_cast<const float4*>(p + 20);
    v16h a;
    a[0]=(_Float16)f0.x; a[1]=(_Float16)f0.y; a[2]=(_Float16)f0.z; a[3]=(_Float16)f0.w;
    a[4]=(_Float16)f1.x; a[5]=(_Float16)f1.y; a[6]=(_Float16)f1.z; a[7]=(_Float16)f1.w;
    a[8]=(_Float16)f2.x; a[9]=(_Float16)f2.y; a[10]=(_Float16)f2.z; a[11]=(_Float16)f2.w;
    a[12]=(_Float16)f3.x; a[13]=(_Float16)f3.y; a[14]=(_Float16)f3.z; a[15]=(_Float16)f3.w;
    return a;
}

// B (32x16 f16): lane L -> N = L&15; half h -> K = kb + (h/8)*16 + group*8 + (h%8)
__device__ __forceinline__ v16h load_b_frag(const float* __restrict__ W, int ldb,
                                            int kb, int group, int colbase) {
    v16h b;
#pragma unroll
    for (int h = 0; h < 16; ++h) {
        int k = kb + ((h >> 3) << 4) + (group << 3) + (h & 7);
        b[h] = (_Float16)W[(size_t)k * ldb + colbase];
    }
    return b;
}

// ---------------------------------------------------------------------------
// Node GEMM:  C[M x Nc] = act(A[M x K] @ W[K x Nc] + bias)
// One 16x16 output tile per wave, 8 waves / 256-thread block.
// ---------------------------------------------------------------------------
__global__ void gemm_wmma_kernel(const float* __restrict__ A, const float* __restrict__ W,
                                 const float* __restrict__ bias, float* __restrict__ C,
                                 int M, int K, int Nc, int relu) {
    const int lane  = threadIdx.x & 31;
    const int wave  = threadIdx.x >> 5;
    const int nTN   = Nc >> 4;
    int tile  = blockIdx.x * 8 + wave;
    int tileM = tile / nTN;
    int tileN = tile - tileM * nTN;
    int m0 = tileM << 4;
    if (m0 >= M) return;

    const int group = lane >> 4;
    const int mloc  = lane & 15;
    const float* arow = A + (size_t)(m0 + mloc) * K + (group << 3);
    const int colbase = (tileN << 4) + mloc;

    v8f acc = {};
    for (int kb = 0; kb < K; kb += 32) {
        v16h a = load_a_frag(arow + kb);
        v16h b = load_b_frag(W, Nc, kb, group, colbase);
        acc = WMMA_F16(a, b, acc);
    }
    float bv = bias ? bias[colbase] : 0.0f;
#pragma unroll
    for (int r = 0; r < 8; ++r) {
        int m = m0 + (group << 3) + r;
        float v = acc[r] + bv;
        if (relu) v = fmaxf(v, 0.0f);
        C[(size_t)m * Nc + colbase] = v;
    }
}

// ---------------------------------------------------------------------------
// Attention edge MLP: score[e] = w2 . relu([h[row],h[col]] @ W1 + b1) + b2
// K = 256 (gathered), mid = 64. One 16-edge tile per wave, 4 col tiles kept
// in registers, then 64->1 layer via per-group shfl_xor reduction.
// ---------------------------------------------------------------------------
__global__ void edge_mlp_wmma_kernel(const float* __restrict__ h,
                                     const int* __restrict__ row, const int* __restrict__ col,
                                     const float* __restrict__ w1, const float* __restrict__ b1,
                                     const float* __restrict__ w2, const float* __restrict__ b2,
                                     float* __restrict__ score, int E) {
    const int lane = threadIdx.x & 31;
    const int wave = threadIdx.x >> 5;
    int tileE = blockIdx.x * 8 + wave;
    int e0 = tileE << 4;
    if (e0 >= E) return;

    const int group = lane >> 4;
    const int mloc  = lane & 15;
    const int e     = e0 + mloc;
    const float* hr = h + (size_t)row[e] * 128;
    const float* hc = h + (size_t)col[e] * 128;

    v8f a0 = {}, a1 = {}, a2 = {}, a3 = {};
    for (int kb = 0; kb < 256; kb += 32) {
        const float* src = (kb < 128) ? (hr + kb) : (hc + kb - 128);
        v16h af = load_a_frag(src + (group << 3));
        v16h bf;
        bf = load_b_frag(w1, 64, kb, group,  0 + mloc); a0 = WMMA_F16(af, bf, a0);
        bf = load_b_frag(w1, 64, kb, group, 16 + mloc); a1 = WMMA_F16(af, bf, a1);
        bf = load_b_frag(w1, 64, kb, group, 32 + mloc); a2 = WMMA_F16(af, bf, a2);
        bf = load_b_frag(w1, 64, kb, group, 48 + mloc); a3 = WMMA_F16(af, bf, a3);
    }
    float out[8];
#pragma unroll
    for (int r = 0; r < 8; ++r) {
        float p = 0.0f;
        p += fmaxf(a0[r] + b1[ 0 + mloc], 0.0f) * w2[ 0 + mloc];
        p += fmaxf(a1[r] + b1[16 + mloc], 0.0f) * w2[16 + mloc];
        p += fmaxf(a2[r] + b1[32 + mloc], 0.0f) * w2[32 + mloc];
        p += fmaxf(a3[r] + b1[48 + mloc], 0.0f) * w2[48 + mloc];
        p += __shfl_xor(p, 1, 32);
        p += __shfl_xor(p, 2, 32);
        p += __shfl_xor(p, 4, 32);
        p += __shfl_xor(p, 8, 32);
        out[r] = p;
    }
    if (mloc == 0) {
#pragma unroll
        for (int r = 0; r < 8; ++r)
            score[e0 + (group << 3) + r] = out[r] + b2[0];
    }
}

// ---------------------------------------------------------------------------
// Edge gather-scatter: dst[row[e]] += src[col[e]] * s   (s = scale[e] or 1)
// float4 per thread, 32 threads per edge.
// ---------------------------------------------------------------------------
__global__ void scatter_kernel(float* __restrict__ dst, const float* __restrict__ src,
                               const int* __restrict__ row, const int* __restrict__ col,
                               const float* __restrict__ scale, int E) {
    int idx = blockIdx.x * blockDim.x + threadIdx.x;
    int e = idx >> 5;
    if (e >= E) return;
    int f = (idx & 31) << 2;
    float s = scale ? scale[e] : 1.0f;
    float4 v = *reinterpret_cast<const float4*>(src + (size_t)col[e] * 128 + f);
    float* d = dst + (size_t)row[e] * 128 + f;
    atomicAdd(d + 0, v.x * s); atomicAdd(d + 1, v.y * s);
    atomicAdd(d + 2, v.z * s); atomicAdd(d + 3, v.w * s);
}

// attention-weighted scatter using global softmax stats in red[6] (max key) / red[7] (sumexp)
__device__ __forceinline__ unsigned enc_ord(float f) {
    unsigned u = __float_as_uint(f);
    return (u & 0x80000000u) ? ~u : (u | 0x80000000u);
}
__device__ __forceinline__ float dec_ord(unsigned k) {
    return __uint_as_float((k & 0x80000000u) ? (k ^ 0x80000000u) : ~k);
}

__global__ void attn_scatter_kernel(float* __restrict__ dst, const float* __restrict__ src,
                                    const int* __restrict__ row, const int* __restrict__ col,
                                    const float* __restrict__ score, const float* __restrict__ red,
                                    int E) {
    int idx = blockIdx.x * blockDim.x + threadIdx.x;
    int e = idx >> 5;
    if (e >= E) return;
    int f = (idx & 31) << 2;
    float m   = dec_ord(__float_as_uint(red[6]));
    float inv = 1.0f / red[7];
    float s = __expf(score[e] - m) * inv;
    float4 v = *reinterpret_cast<const float4*>(src + (size_t)col[e] * 128 + f);
    float* d = dst + (size_t)row[e] * 128 + f;
    atomicAdd(d + 0, v.x * s); atomicAdd(d + 1, v.y * s);
    atomicAdd(d + 2, v.z * s); atomicAdd(d + 3, v.w * s);
}

// ---------------------------------------------------------------------------
// Reductions
// ---------------------------------------------------------------------------
__global__ void deg_kernel(const int* __restrict__ row, const float* __restrict__ ew,
                           float* __restrict__ deg, float* __restrict__ degw, int E) {
    int e = blockIdx.x * blockDim.x + threadIdx.x;
    if (e >= E) return;
    atomicAdd(&deg[row[e]], 1.0f);
    atomicAdd(&degw[row[e]], ew[e]);
}

__global__ void dis_kernel(const float* __restrict__ deg, float* __restrict__ dis, int N) {
    int i = blockIdx.x * blockDim.x + threadIdx.x;
    if (i >= N) return;
    dis[i] = fmaxf(rsqrtf(fmaxf(deg[i], 1e-30f)), 1e-8f);
}

__global__ void norm_edge_kernel(const float* __restrict__ dis, const int* __restrict__ row,
                                 const int* __restrict__ col, float* __restrict__ nrm, int E) {
    int e = blockIdx.x * blockDim.x + threadIdx.x;
    if (e >= E) return;
    nrm[e] = dis[row[e]] * dis[col[e]];
}

// mode 0: deg -> red[0..2] (sum, sumsq, #zero); mode 1: x -> red[3..4]; mode 2: ew -> red[5]
__global__ void reduce_stats_kernel(const float* __restrict__ a, int n,
                                    float* __restrict__ red, int mode) {
    float s = 0.f, s2 = 0.f, z = 0.f;
    for (int i = blockIdx.x * blockDim.x + threadIdx.x; i < n; i += gridDim.x * blockDim.x) {
        float v = a[i];
        s += v; s2 += v * v; z += (v == 0.0f) ? 1.0f : 0.0f;
    }
    __shared__ float sh[3][8];
    for (int o = 16; o > 0; o >>= 1) {
        s += __shfl_down(s, o); s2 += __shfl_down(s2, o); z += __shfl_down(z, o);
    }
    int wave = threadIdx.x >> 5, lane = threadIdx.x & 31;
    if (lane == 0) { sh[0][wave] = s; sh[1][wave] = s2; sh[2][wave] = z; }
    __syncthreads();
    if (threadIdx.x == 0) {
        float S = 0.f, S2 = 0.f, Z = 0.f;
        int nw = blockDim.x >> 5;
        for (int w = 0; w < nw; ++w) { S += sh[0][w]; S2 += sh[1][w]; Z += sh[2][w]; }
        if (mode == 0) { atomicAdd(&red[0], S); atomicAdd(&red[1], S2); atomicAdd(&red[2], Z); }
        else if (mode == 1) { atomicAdd(&red[3], S); atomicAdd(&red[4], S2); }
        else { atomicAdd(&red[5], S); }
    }
}

__global__ void softmax_init_kernel(float* __restrict__ red) {
    if (threadIdx.x == 0 && blockIdx.x == 0) {
        ((unsigned*)red)[6] = 0x007FFFFFu;   // enc(-inf)
        red[7] = 0.0f;
    }
}

__global__ void score_max_kernel(const float* __restrict__ score, int E, float* __restrict__ red) {
    unsigned k = 0x007FFFFFu;
    for (int i = blockIdx.x * blockDim.x + threadIdx.x; i < E; i += gridDim.x * blockDim.x) {
        unsigned c = enc_ord(score[i]);
        k = (c > k) ? c : k;
    }
    for (int o = 16; o > 0; o >>= 1) {
        unsigned c = (unsigned)__shfl_down((int)k, o);
        k = (c > k) ? c : k;
    }
    __shared__ unsigned sm[8];
    int wave = threadIdx.x >> 5, lane = threadIdx.x & 31;
    if (lane == 0) sm[wave] = k;
    __syncthreads();
    if (threadIdx.x == 0) {
        unsigned K = 0x007FFFFFu;
        int nw = blockDim.x >> 5;
        for (int w = 0; w < nw; ++w) K = (sm[w] > K) ? sm[w] : K;
        atomicMax((unsigned*)red + 6, K);
    }
}

__global__ void score_sumexp_kernel(const float* __restrict__ score, int E, float* __restrict__ red) {
    float m = dec_ord(((const unsigned*)red)[6]);
    float s = 0.0f;
    for (int i = blockIdx.x * blockDim.x + threadIdx.x; i < E; i += gridDim.x * blockDim.x)
        s += __expf(score[i] - m);
    for (int o = 16; o > 0; o >>= 1) s += __shfl_down(s, o);
    __shared__ float sm[8];
    int wave = threadIdx.x >> 5, lane = threadIdx.x & 31;
    if (lane == 0) sm[wave] = s;
    __syncthreads();
    if (threadIdx.x == 0) {
        float S = 0.0f;
        int nw = blockDim.x >> 5;
        for (int w = 0; w < nw; ++w) S += sm[w];
        atomicAdd(&red[7], S);
    }
}

// ---------------------------------------------------------------------------
// Controller MLP + gumbel softmax gating (single block)
// ---------------------------------------------------------------------------
__global__ void controller_kernel(const float* __restrict__ red, const float* __restrict__ prev,
                                  const float* __restrict__ gum,
                                  const float* __restrict__ w1, const float* __restrict__ b1,
                                  const float* __restrict__ w2, const float* __restrict__ b2,
                                  const float* __restrict__ mode_bias,
                                  const float* __restrict__ scale_w,
                                  float* __restrict__ probs, int N, int E) {
    __shared__ float s[136];
    __shared__ float hid[128];
    __shared__ float logit[5];
    int t = threadIdx.x;
    if (t == 0) {
        float n = (float)N, e = (float)E;
        float sd = red[0], sd2 = red[1], z = red[2], sx = red[3], sx2 = red[4], sew = red[5];
        s[0] = n / 1000.0f;
        s[1] = e / n;
        s[2] = sqrtf(fmaxf((sd2 - sd * sd / n) / (n - 1.0f), 0.0f));
        s[3] = z / n;
        float M = n * 128.0f;
        s[4] = sx / M;
        s[5] = sqrtf(fmaxf((sx2 - sx * sx / M) / (M - 1.0f), 0.0f));
        s[6] = sew / e;
        s[7] = e / (n * n);
    }
    if (t < 128) s[8 + t] = prev[t];
    __syncthreads();
    if (t < 128) {
        float a = b1[t];
        for (int j = 0; j < 136; ++j) a += s[j] * w1[j * 128 + t];
        hid[t] = fmaxf(a, 0.0f);
    }
    __syncthreads();
    if (t < 5) {
        float a = b2[t] + mode_bias[t];
        for (int j = 0; j < 128; ++j) a += hid[j] * w2[j * 5 + t];
        logit[t] = (a + gum[t]) * 2.0f;   // divide by TAU = 0.5
    }
    __syncthreads();
    if (t == 0) {
        float mx = logit[0];
        for (int i = 1; i < 5; ++i) mx = fmaxf(mx, logit[i]);
        float p[5], den = 0.0f;
        for (int i = 0; i < 5; ++i) { p[i] = __expf(logit[i] - mx); den += p[i]; }
        for (int i = 0; i < 5; ++i) {
            float q = p[i] / den;
            probs[i] = (q > 0.001f) ? q : 0.0f;
        }
        float m2 = fmaxf(fmaxf(scale_w[0], scale_w[1]), scale_w[2]);
        float e0 = __expf(scale_w[0] - m2), e1 = __expf(scale_w[1] - m2), e2 = __expf(scale_w[2] - m2);
        float d2 = e0 + e1 + e2;
        probs[5] = e0 / d2; probs[6] = e1 / d2; probs[7] = e2 / d2;
    }
}

// ---------------------------------------------------------------------------
// Elementwise epilogues
// ---------------------------------------------------------------------------
__global__ void divdeg_kernel(float* __restrict__ x, const float* __restrict__ deg,
                              int NH, int relu) {
    int i = blockIdx.x * blockDim.x + threadIdx.x;
    if (i >= NH) return;
    float v = x[i] / fmaxf(deg[i >> 7], 1.0f);
    x[i] = relu ? fmaxf(v, 0.0f) : v;
}

__global__ void temporal_combine_kernel(float* __restrict__ dst, const float* __restrict__ h,
                                        const float* __restrict__ g0, const float* __restrict__ agg,
                                        const float* __restrict__ ts, const float* __restrict__ tg_last,
                                        int NH) {
    int i = blockIdx.x * blockDim.x + threadIdx.x;
    if (i >= NH) return;
    float gate = 1.0f / (1.0f + __expf(-(g0[i] + ts[i >> 7] * tg_last[i & 127])));
    dst[i] = fmaxf(gate * h[i] + (1.0f - gate) * agg[i], 0.0f);
}

__global__ void relu_copy_kernel(float* __restrict__ dst, const float* __restrict__ src, int n) {
    int i = blockIdx.x * blockDim.x + threadIdx.x;
    if (i >= n) return;
    dst[i] = fmaxf(src[i], 0.0f);
}

__global__ void heat_step_kernel(float* __restrict__ h, const float* __restrict__ hk,
                                 float* __restrict__ acc, const float* __restrict__ heat,
                                 int j, int n) {
    int i = blockIdx.x * blockDim.x + threadIdx.x;
    if (i >= n) return;
    float t = heat[j];
    float v = (1.0f - t) * h[i] + t * hk[i];
    h[i] = v;
    acc[i] += v;
}

__global__ void diff_final_kernel(float* __restrict__ dst, const float* __restrict__ acc, int n) {
    int i = blockIdx.x * blockDim.x + threadIdx.x;
    if (i >= n) return;
    dst[i] = fmaxf(acc[i] * 0.2f, 0.0f);
}

__global__ void hier_combine_kernel(float* __restrict__ dst, const float* __restrict__ h1,
                                    const float* __restrict__ h2, const float* __restrict__ h3,
                                    const float* __restrict__ probs, int n) {
    int i = blockIdx.x * blockDim.x + threadIdx.x;
    if (i >= n) return;
    dst[i] = fmaxf(probs[5] * h1[i] + probs[6] * h2[i] + probs[7] * h3[i], 0.0f);
}

__global__ void accum_kernel(float* __restrict__ xsum, const float* __restrict__ src,
                             const float* __restrict__ probs, int gi, int n) {
    int i = blockIdx.x * blockDim.x + threadIdx.x;
    if (i >= n) return;
    xsum[i] += probs[gi] * src[i];
}

// ---------------------------------------------------------------------------
// Host orchestration
// ---------------------------------------------------------------------------
extern "C" void kernel_launch(void* const* d_in, const int* in_sizes, int n_in,
                              void* d_out, int out_size, void* d_ws, size_t ws_size,
                              hipStream_t stream) {
    const int H = 128;
    const int N = in_sizes[0] / H;
    const int E = in_sizes[3];
    const int NH = N * H;

    const float* x        = (const float*)d_in[0];
    const int*   row      = (const int*)d_in[1];
    const int*   col      = row + E;
    const float* ts       = (const float*)d_in[2];
    const float* ew       = (const float*)d_in[3];
    const float* prev     = (const float*)d_in[4];
    const float* gum      = (const float*)d_in[5];
    const float* ctrl_w1  = (const float*)d_in[6];
    const float* ctrl_b1  = (const float*)d_in[7];
    const float* ctrl_w2  = (const float*)d_in[8];
    const float* ctrl_b2  = (const float*)d_in[9];
    const float* mode_b   = (const float*)d_in[10];
    const float* att_w1   = (const float*)d_in[11];
    const float* att_b1   = (const float*)d_in[12];
    const float* att_w2   = (const float*)d_in[13];
    const float* att_b2   = (const float*)d_in[14];
    const float* heat     = (const float*)d_in[15];
    const float* tg_w     = (const float*)d_in[16];
    const float* tg_b     = (const float*)d_in[17];
    const float* scale_w  = (const float*)d_in[18];
    const float* sp_w     = (const float*)d_in[19];
    const float* sp_b     = (const float*)d_in[20];
    const float* tp_w     = (const float*)d_in[21];
    const float* tp_b     = (const float*)d_in[22];
    const float* at_w     = (const float*)d_in[23];
    const float* at_b     = (const float*)d_in[24];
    const float* df_w     = (const float*)d_in[25];
    const float* df_b     = (const float*)d_in[26];
    const float* hi_w     = (const float*)d_in[27];
    const float* hi_b     = (const float*)d_in[28];
    const float* out_w1   = (const float*)d_in[29];
    const float* out_b1   = (const float*)d_in[30];
    const float* out_w2   = (const float*)d_in[31];
    const float* out_b2   = (const float*)d_in[32];
    float* out = (float*)d_out;

    const size_t F = (size_t)NH;
    float* w     = (float*)d_ws;
    float* cur   = w;
    float* hbuf  = w + 1 * F;
    float* agg   = w + 2 * F;
    float* t3    = w + 3 * F;
    float* t4    = w + 4 * F;
    float* xsum  = w + 5 * F;
    float* mid   = w + 6 * F;                    // N x 64
    float* deg   = mid + (size_t)N * 64;
    float* degw  = deg + N;
    float* dis   = degw + N;
    float* score = dis + N;                      // E floats
    float* red   = score + E;                    // 16 floats
    float* probs = red + 16;                     // 8 floats

    const int TB = 256;
    const int ebNH = (NH + TB - 1) / TB;
    const int ebE  = (E + TB - 1) / TB;
    const int ebSc = (E * 32 + TB - 1) / TB;     // scatter: 32 threads / edge
    const size_t FB = F * sizeof(float);

    auto gemm = [&](const float* A, const float* W, const float* B, float* C,
                    int M, int K, int Nc, int relu) {
        int tiles = (M / 16) * (Nc / 16);
        gemm_wmma_kernel<<<(tiles + 7) / 8, TB, 0, stream>>>(A, W, B, C, M, K, Nc, relu);
    };

    // --- setup: degrees + stats + controller ---
    hipMemsetAsync(deg, 0, 2 * (size_t)N * sizeof(float), stream);
    hipMemsetAsync(red, 0, 24 * sizeof(float), stream);
    deg_kernel<<<ebE, TB, 0, stream>>>(row, ew, deg, degw, E);
    dis_kernel<<<(N + TB - 1) / TB, TB, 0, stream>>>(deg, dis, N);
    reduce_stats_kernel<<<512, TB, 0, stream>>>(deg, N, red, 0);
    reduce_stats_kernel<<<1024, TB, 0, stream>>>(x, NH, red, 1);
    reduce_stats_kernel<<<512, TB, 0, stream>>>(ew, E, red, 2);
    controller_kernel<<<1, 128, 0, stream>>>(red, prev, gum, ctrl_w1, ctrl_b1,
                                             ctrl_w2, ctrl_b2, mode_b, scale_w,
                                             probs, N, E);
    hipMemsetAsync(xsum, 0, FB, stream);

    // --- pass 0: spatial ---
    gemm(x, sp_w, sp_b, hbuf, N, H, H, 0);
    hipMemsetAsync(agg, 0, FB, stream);
    scatter_kernel<<<ebSc, TB, 0, stream>>>(agg, hbuf, row, col, ew, E);
    divdeg_kernel<<<ebNH, TB, 0, stream>>>(agg, degw, NH, 1);
    gemm(agg, sp_w + H * H, sp_b + H, hbuf, N, H, H, 0);
    hipMemsetAsync(cur, 0, FB, stream);
    scatter_kernel<<<ebSc, TB, 0, stream>>>(cur, hbuf, row, col, ew, E);
    divdeg_kernel<<<ebNH, TB, 0, stream>>>(cur, degw, NH, 1);
    accum_kernel<<<ebNH, TB, 0, stream>>>(xsum, cur, probs, 0, NH);

    // --- pass 1: temporal ---
    const float* tg_last = tg_w + (size_t)H * H;   // row 128 of (129,128)
    for (int l = 0; l < 2; ++l) {
        const float* in = (l == 0) ? x : cur;
        gemm(in, tp_w + (size_t)l * H * H, tp_b + (size_t)l * H, hbuf, N, H, H, 0);
        gemm(hbuf, tg_w, tg_b, t3, N, H, H, 0);
        hipMemsetAsync(agg, 0, FB, stream);
        scatter_kernel<<<ebSc, TB, 0, stream>>>(agg, hbuf, row, col, nullptr, E);
        divdeg_kernel<<<ebNH, TB, 0, stream>>>(agg, deg, NH, 0);
        temporal_combine_kernel<<<ebNH, TB, 0, stream>>>(cur, hbuf, t3, agg, ts, tg_last, NH);
    }
    accum_kernel<<<ebNH, TB, 0, stream>>>(xsum, cur, probs, 1, NH);

    // --- pass 2: attention ---
    for (int l = 0; l < 2; ++l) {
        const float* in = (l == 0) ? x : cur;
        gemm(in, at_w + (size_t)l * H * H, at_b + (size_t)l * H, hbuf, N, H, H, 0);
        edge_mlp_wmma_kernel<<<(E / 16 + 7) / 8, TB, 0, stream>>>(
            hbuf, row, col, att_w1, att_b1, att_w2, att_b2, score, E);
        softmax_init_kernel<<<1, 32, 0, stream>>>(red);
        score_max_kernel<<<512, TB, 0, stream>>>(score, E, red);
        score_sumexp_kernel<<<512, TB, 0, stream>>>(score, E, red);
        hipMemsetAsync(agg, 0, FB, stream);
        attn_scatter_kernel<<<ebSc, TB, 0, stream>>>(agg, hbuf, row, col, score, red, E);
        relu_copy_kernel<<<ebNH, TB, 0, stream>>>(cur, agg, NH);
    }
    accum_kernel<<<ebNH, TB, 0, stream>>>(xsum, cur, probs, 2, NH);

    // --- pass 3: diffusion ---
    norm_edge_kernel<<<ebE, TB, 0, stream>>>(dis, row, col, score, E);
    for (int l = 0; l < 2; ++l) {
        const float* in = (l == 0) ? x : t3;
        gemm(in, df_w + (size_t)l * H * H, df_b + (size_t)l * H, hbuf, N, H, H, 0);
        hipMemsetAsync(t4, 0, FB, stream);
        for (int j = 0; j < 5; ++j) {
            hipMemsetAsync(agg, 0, FB, stream);
            scatter_kernel<<<ebSc, TB, 0, stream>>>(agg, hbuf, row, col, score, E);
            heat_step_kernel<<<ebNH, TB, 0, stream>>>(hbuf, agg, t4, heat, j, NH);
        }
        diff_final_kernel<<<ebNH, TB, 0, stream>>>((l == 0) ? t3 : cur, t4, NH);
    }
    accum_kernel<<<ebNH, TB, 0, stream>>>(xsum, cur, probs, 3, NH);

    // --- pass 4: hierarchical ---
    for (int l = 0; l < 2; ++l) {
        const float* in = (l == 0) ? x : cur;
        gemm(in, hi_w + (size_t)l * H * H, hi_b + (size_t)l * H, hbuf, N, H, H, 0);
        hipMemsetAsync(agg, 0, FB, stream);
        scatter_kernel<<<ebSc, TB, 0, stream>>>(agg, hbuf, row, col, nullptr, E);
        divdeg_kernel<<<ebNH, TB, 0, stream>>>(agg, deg, NH, 0);
        hipMemsetAsync(t3, 0, FB, stream);
        scatter_kernel<<<ebSc, TB, 0, stream>>>(t3, agg, row, col, nullptr, E);
        divdeg_kernel<<<ebNH, TB, 0, stream>>>(t3, deg, NH, 0);
        hipMemsetAsync(t4, 0, FB, stream);
        scatter_kernel<<<ebSc, TB, 0, stream>>>(t4, t3, row, col, nullptr, E);
        divdeg_kernel<<<ebNH, TB, 0, stream>>>(t4, deg, NH, 0);
        hier_combine_kernel<<<ebNH, TB, 0, stream>>>(cur, agg, t3, t4, probs, NH);
    }
    accum_kernel<<<ebNH, TB, 0, stream>>>(xsum, cur, probs, 4, NH);

    // --- output MLP: 128 -> 64 (relu) -> 64 ---
    gemm(xsum, out_w1, out_b1, mid, N, 128, 64, 1);
    gemm(mid, out_w2, out_b2, out, N, 64, 64, 0);
}
